// RelativeMultiHeadAttention_16690242912704
// MI455X (gfx1250) — compile-verified, matrix operand order checked by reading, not run on hardware
//
#include <hip/hip_runtime.h>
#include <hip/hip_bf16.h>

// ---------------------------------------------------------------------------
// Transformer-XL relative multi-head attention for MI455X (gfx1250, wave32)
// All matmuls via v_wmma_f32_16x16x32_bf16. Intermediates bf16 in d_ws:
//   qh | kh | vh | ph | ctx  : 5 x 16 MiB = 80 MiB
// bf16 tile staging uses GLOBAL_LOAD_ASYNC_TO_LDS_B128 (+ s_wait_asynccnt).
// ---------------------------------------------------------------------------

typedef __attribute__((ext_vector_type(16))) __bf16 bfx16;
typedef __attribute__((ext_vector_type(8)))  float  fx8;
typedef __attribute__((ext_vector_type(4)))  int    i32x4_;

#define DMODEL 1024
#define SEQ    1024
#define NHEAD  16
#define DHEAD  64

#if __has_builtin(__builtin_amdgcn_global_load_async_to_lds_b128) && \
    __has_builtin(__builtin_amdgcn_s_wait_asynccnt)
#define HAVE_ASYNC_COPY 1
#else
#define HAVE_ASYNC_COPY 0
#endif

#if HAVE_ASYNC_COPY
typedef __attribute__((address_space(1))) i32x4_* g128p;
typedef __attribute__((address_space(3))) i32x4_* l128p;
typedef __attribute__((address_space(1))) void*   gvoidp;
typedef __attribute__((address_space(3))) void*   lvoidp;

// async copy of one 16-byte chunk global -> LDS (ASYNCcnt tracked)
__device__ __forceinline__ void async_cp16(const void* g, void* l) {
    __builtin_amdgcn_global_load_async_to_lds_b128(
        (g128p)(gvoidp)g, (l128p)(lvoidp)l, 0, 0);
}
#endif

union BF16Frag { bfx16 v; __bf16 a[16]; };

// per-lane K index of element e of a 16-bit 16x32 A/B fragment (wave32)
__device__ __forceinline__ int kofs(int e, int hl) {
    return ((e >> 3) << 4) + hl * 8 + (e & 7);
}

__device__ __forceinline__ fx8 fzero8() {
    fx8 z = {0.f, 0.f, 0.f, 0.f, 0.f, 0.f, 0.f, 0.f};
    return z;
}

// Stage a 64x64 bf16 tile (global row stride = DMODEL elements) into
// contiguous LDS [64][64]. 8192 B total; 256 threads.
__device__ __forceinline__ void stage_tile_64x64(const __bf16* __restrict__ g,
                                                 __bf16* __restrict__ l, int tid) {
#if HAVE_ASYNC_COPY
    #pragma unroll
    for (int i = 0; i < 2; ++i) {
        int c = tid + i * 256;              // 16-byte chunk id, 0..511
        int r = c >> 3, col = (c & 7) * 8;  // 8 chunks per 64-elem row
        async_cp16(g + (size_t)r * DMODEL + col, l + r * 64 + col);
    }
    __builtin_amdgcn_s_wait_asynccnt(0);    // per-wave; caller barriers after
#else
    #pragma unroll
    for (int i = 0; i < 16; ++i) {
        int ee = tid + i * 256;
        int r = ee >> 6, c = ee & 63;
        l[r * 64 + c] = g[(size_t)r * DMODEL + c];
    }
#endif
}

// ---------------------------------------------------------------------------
// C[M,N] = A[M,K] @ W[N,K]^T (+ bias[N])   (torch Linear semantics)
// block: 256 threads (8 waves). Tile 128(M) x 64(N), BK=32.
// Wave grid 4x2, each wave owns a 32x32 output tile (2x2 WMMA tiles).
// ---------------------------------------------------------------------------
template <bool A_BF16, bool OUT_BF16>
__global__ __launch_bounds__(256) void gemm_xwt_kernel(
    const void* __restrict__ Av, const float* __restrict__ W,
    const float* __restrict__ bias, void* __restrict__ Cv,
    int M, int N, int K)
{
    constexpr int BM = 128, BN = 64, BK = 32, LDA = BK + 8;
    __shared__ __bf16 As[BM * LDA];
    __shared__ __bf16 Bs[BN * LDA];

    const int tid  = threadIdx.x;
    const int wave = tid >> 5;
    const int lane = tid & 31;
    const int hl   = lane >> 4;
    const int l16  = lane & 15;

    const int n0 = blockIdx.x * BN;
    const int m0 = blockIdx.y * BM;

    const int wr = wave & 3;   // 32-row strip within block tile
    const int wc = wave >> 2;  // 32-col strip within block tile

    const float*  Af = (const float*)Av;
    const __bf16* Ab = (const __bf16*)Av;

    fx8 acc[4];
    #pragma unroll
    for (int t = 0; t < 4; ++t) acc[t] = fzero8();

    for (int k0 = 0; k0 < K; k0 += BK) {
        // stage A tile 128x32 -> bf16 LDS
        if (A_BF16) {
#if HAVE_ASYNC_COPY
            #pragma unroll
            for (int i = 0; i < 2; ++i) {
                int c = tid + i * 256;              // 512 x 16-byte chunks
                int r = c >> 2, col = (c & 3) * 8;  // 4 chunks per 32-elem row
                async_cp16(Ab + (size_t)(m0 + r) * K + k0 + col,
                           &As[r * LDA + col]);
            }
#else
            #pragma unroll
            for (int i = 0; i < 4; ++i) {
                int r  = (tid >> 3) + i * 32;
                int c4 = (tid & 7) * 4;
                #pragma unroll
                for (int j = 0; j < 4; ++j)
                    As[r * LDA + c4 + j] = Ab[(size_t)(m0 + r) * K + k0 + c4 + j];
            }
#endif
        } else {
            #pragma unroll
            for (int i = 0; i < 4; ++i) {
                int r  = (tid >> 3) + i * 32;
                int c4 = (tid & 7) * 4;
                float4 vv = *(const float4*)&Af[(size_t)(m0 + r) * K + k0 + c4];
                As[r * LDA + c4 + 0] = (__bf16)vv.x;
                As[r * LDA + c4 + 1] = (__bf16)vv.y;
                As[r * LDA + c4 + 2] = (__bf16)vv.z;
                As[r * LDA + c4 + 3] = (__bf16)vv.w;
            }
        }
        // stage W tile 64x32 -> bf16 LDS (fp32 source: convert through VGPRs)
        #pragma unroll
        for (int i = 0; i < 2; ++i) {
            int r  = (tid >> 3) + i * 32;
            int c4 = (tid & 7) * 4;
            float4 vv = *(const float4*)&W[(size_t)(n0 + r) * K + k0 + c4];
            Bs[r * LDA + c4 + 0] = (__bf16)vv.x;
            Bs[r * LDA + c4 + 1] = (__bf16)vv.y;
            Bs[r * LDA + c4 + 2] = (__bf16)vv.z;
            Bs[r * LDA + c4 + 3] = (__bf16)vv.w;
        }
        if (k0 + BK < K) {  // global_prefetch_b8 of next K-slab
            if (!A_BF16)
                __builtin_prefetch(&Af[(size_t)(m0 + (tid >> 3)) * K + k0 + BK], 0, 1);
            __builtin_prefetch(&W[(size_t)(n0 + (tid >> 3)) * K + k0 + BK], 0, 1);
        }
#if HAVE_ASYNC_COPY
        if (A_BF16) __builtin_amdgcn_s_wait_asynccnt(0);
#endif
        __syncthreads();

        BF16Frag afr[2], bfr[2];
        #pragma unroll
        for (int t = 0; t < 2; ++t) {
            #pragma unroll
            for (int e = 0; e < 16; ++e) {
                afr[t].a[e] = As[(wr * 32 + t * 16 + l16) * LDA + kofs(e, hl)];
                bfr[t].a[e] = Bs[(wc * 32 + t * 16 + l16) * LDA + kofs(e, hl)];
            }
        }
        #pragma unroll
        for (int t = 0; t < 2; ++t)
            #pragma unroll
            for (int u = 0; u < 2; ++u)
                acc[t * 2 + u] = __builtin_amdgcn_wmma_f32_16x16x32_bf16(
                    false, afr[t].v, false, bfr[u].v, (short)0, acc[t * 2 + u],
                    false, false);
        __syncthreads();
    }

    #pragma unroll
    for (int t = 0; t < 2; ++t)
        #pragma unroll
        for (int u = 0; u < 2; ++u)
            #pragma unroll
            for (int r = 0; r < 8; ++r) {
                int gm = m0 + wr * 32 + t * 16 + r + hl * 8;
                int gn = n0 + wc * 32 + u * 16 + l16;
                float vv = acc[t * 2 + u][r];
                if (bias) vv += bias[gn];
                if (OUT_BF16) ((__bf16*)Cv)[(size_t)gm * N + gn] = (__bf16)vv;
                else          ((float*)Cv)[(size_t)gm * N + gn]  = vv;
            }
}

// ---------------------------------------------------------------------------
// Fused TXL attention for one (b, h, 64-row i-block).
// Rel-shift identity: shifted_pos[i,j] = P[i, j-i+S-1] if j<=i else 0,
// where P = (q+v_bias) @ ph^T. Each P element is consumed at most once, so we
// precompute the full 64x1024 P strip into LDS (256 KB -- CDNA5 WGP LDS),
// then run an online-softmax flash loop over 64-wide j blocks.
// Dynamic LDS layout (bytes):
//   Pbuf  f32 [64][1024]   262144
//   SC    f32 [64][64]      16384
//   STG   bf16[64][64]       8192   (kh / ph / vh staging, async-loaded)
//   PB    bf16[64][64]       8192   (softmax probabilities)
//   stats f32 64*3 + 256     1792
// total 296704 B
// ---------------------------------------------------------------------------
#define ATT_SMEM (262144 + 16384 + 8192 + 8192 + 1792)

__global__ __launch_bounds__(256) void txl_attn_kernel(
    const __bf16* __restrict__ qh, const __bf16* __restrict__ kh,
    const __bf16* __restrict__ vh, const __bf16* __restrict__ ph,
    const float* __restrict__ u_bias, const float* __restrict__ v_bias,
    __bf16* __restrict__ ctx)
{
    extern __shared__ char smem[];
    float*  Pbuf = (float*)smem;                              // [64][1024]
    float*  SC   = (float*)(smem + 262144);                   // [64][64]
    __bf16* STG  = (__bf16*)(smem + 262144 + 16384);          // [64][64]
    __bf16* PB   = (__bf16*)(smem + 262144 + 16384 + 8192);   // [64][64]
    float*  mst  = (float*)(smem + 262144 + 16384 + 16384);   // [64]
    float*  lst  = mst + 64;
    float*  csc  = lst + 64;
    float*  part = csc + 64;                                  // [64][4]

    const int tid  = threadIdx.x;
    const int wave = tid >> 5;
    const int lane = tid & 31;
    const int hl   = lane >> 4;
    const int l16  = lane & 15;

    const int b  = blockIdx.z;
    const int h  = blockIdx.y;
    const int i0 = blockIdx.x * 64;

    const int tr  = wave & 3;         // 16-row tile strip (score/P/O)
    const int tcp = (wave >> 2) * 2;  // score/P: col tile pair base
    const int wc  = wave >> 2;        // O: 32-wide dh half

    const size_t base = ((size_t)b * SEQ) * DMODEL + (size_t)h * DHEAD;

    // Build A fragments for (q+u) and (q+v): rows i0+tr*16..+15, K over d(0..63)
    BF16Frag au[2], av[2];
    #pragma unroll
    for (int s = 0; s < 2; ++s)
        #pragma unroll
        for (int e = 0; e < 16; ++e) {
            int d   = kofs(e, hl) + 32 * s;
            int row = i0 + tr * 16 + l16;
            float qv = (float)qh[base + (size_t)row * DMODEL + d];
            au[s].a[e] = (__bf16)(qv + u_bias[h * DHEAD + d]);
            av[s].a[e] = (__bf16)(qv + v_bias[h * DHEAD + d]);
        }

    if (tid < 64) { mst[tid] = -3.0e38f; lst[tid] = 0.0f; }

    // ---- Stage 1: P[64][1024] = (q+v_bias) @ ph^T ------------------------
    for (int mb = 0; mb < 16; ++mb) {
        __syncthreads();
        stage_tile_64x64(ph + base + (size_t)(mb * 64) * DMODEL, STG, tid);
        __syncthreads();
        #pragma unroll
        for (int tt = 0; tt < 2; ++tt) {
            int tc = tcp + tt;
            fx8 a = fzero8();
            #pragma unroll
            for (int s = 0; s < 2; ++s) {
                BF16Frag bf0;
                #pragma unroll
                for (int e = 0; e < 16; ++e)
                    bf0.a[e] = STG[(tc * 16 + l16) * 64 + kofs(e, hl) + 32 * s];
                a = __builtin_amdgcn_wmma_f32_16x16x32_bf16(
                    false, av[s].v, false, bf0.v, (short)0, a, false, false);
            }
            #pragma unroll
            for (int r = 0; r < 8; ++r)
                Pbuf[(tr * 16 + r + hl * 8) * 1024 + mb * 64 + tc * 16 + l16] = a[r];
        }
    }

    // ---- Stage 2: flash loop over j blocks -------------------------------
    fx8 o[2];
    o[0] = fzero8(); o[1] = fzero8();

    for (int jb = 0; jb < 16; ++jb) {
        __syncthreads();  // protects STG + Pbuf(first iter)
        stage_tile_64x64(kh + base + (size_t)(jb * 64) * DMODEL, STG, tid);
        __syncthreads();

        // x-score tiles + shifted pos + scale -> SC
        #pragma unroll
        for (int tt = 0; tt < 2; ++tt) {
            int tc = tcp + tt;
            fx8 a = fzero8();
            #pragma unroll
            for (int s = 0; s < 2; ++s) {
                BF16Frag bf0;
                #pragma unroll
                for (int e = 0; e < 16; ++e)
                    bf0.a[e] = STG[(tc * 16 + l16) * 64 + kofs(e, hl) + 32 * s];
                a = __builtin_amdgcn_wmma_f32_16x16x32_bf16(
                    false, au[s].v, false, bf0.v, (short)0, a, false, false);
            }
            #pragma unroll
            for (int r = 0; r < 8; ++r) {
                int il = tr * 16 + r + hl * 8;
                int jl = tc * 16 + l16;
                int gi = i0 + il, gj = jb * 64 + jl;
                float vv = a[r];
                if (gj <= gi) vv += Pbuf[il * 1024 + (gj - gi + SEQ - 1)];
                SC[il * 64 + jl] = vv * 0.03125f;  // 1/sqrt(1024)
            }
        }
        __syncthreads();

        // online softmax: 4 threads per row
        {
            int r = tid >> 2, q4 = tid & 3;
            float mx = -3.0e38f;
            #pragma unroll
            for (int c = 0; c < 16; ++c)
                mx = fmaxf(mx, SC[r * 64 + q4 * 16 + c]);
            part[r * 4 + q4] = mx;
        }
        __syncthreads();
        if (tid < 64) {
            float pm = fmaxf(fmaxf(part[tid * 4], part[tid * 4 + 1]),
                             fmaxf(part[tid * 4 + 2], part[tid * 4 + 3]));
            float mnew = fmaxf(mst[tid], pm);
            csc[tid] = __expf(mst[tid] - mnew);
            mst[tid] = mnew;
        }
        __syncthreads();
        {
            int r = tid >> 2, q4 = tid & 3;
            float mn = mst[r], sum = 0.f;
            #pragma unroll
            for (int c = 0; c < 16; ++c) {
                float p = __expf(SC[r * 64 + q4 * 16 + c] - mn);
                PB[r * 64 + q4 * 16 + c] = (__bf16)p;
                sum += p;
            }
            part[r * 4 + q4] = sum;
        }
        __syncthreads();
        if (tid < 64)
            lst[tid] = lst[tid] * csc[tid] + part[tid * 4] + part[tid * 4 + 1] +
                       part[tid * 4 + 2] + part[tid * 4 + 3];
        __syncthreads();

        // stage vh block (kh no longer needed)
        stage_tile_64x64(vh + base + (size_t)(jb * 64) * DMODEL, STG, tid);
        __syncthreads();

        // rescale running O, then O += P @ Vblk
        #pragma unroll
        for (int tt = 0; tt < 2; ++tt)
            #pragma unroll
            for (int r = 0; r < 8; ++r)
                o[tt][r] = o[tt][r] * csc[tr * 16 + r + hl * 8];

        #pragma unroll
        for (int s = 0; s < 2; ++s) {
            BF16Frag pa;
            #pragma unroll
            for (int e = 0; e < 16; ++e)
                pa.a[e] = PB[(tr * 16 + l16) * 64 + kofs(e, hl) + 32 * s];
            #pragma unroll
            for (int tt = 0; tt < 2; ++tt) {
                BF16Frag vbf;
                #pragma unroll
                for (int e = 0; e < 16; ++e)
                    vbf.a[e] = STG[(kofs(e, hl) + 32 * s) * 64 + wc * 32 + tt * 16 + l16];
                o[tt] = __builtin_amdgcn_wmma_f32_16x16x32_bf16(
                    false, pa.v, false, vbf.v, (short)0, o[tt], false, false);
            }
        }
    }
    __syncthreads();

    // normalize and write context (bf16)
    #pragma unroll
    for (int tt = 0; tt < 2; ++tt)
        #pragma unroll
        for (int r = 0; r < 8; ++r) {
            int il = tr * 16 + r + hl * 8;
            float inv = 1.0f / lst[il];
            int row = i0 + il;
            int d = wc * 32 + tt * 16 + l16;
            ctx[base + (size_t)row * DMODEL + d] = (__bf16)(o[tt][r] * inv);
        }
}

// ---------------------------------------------------------------------------
extern "C" void kernel_launch(void* const* d_in, const int* in_sizes, int n_in,
                              void* d_out, int out_size, void* d_ws, size_t ws_size,
                              hipStream_t stream) {
    (void)in_sizes; (void)n_in; (void)out_size; (void)ws_size;
    const float* q  = (const float*)d_in[0];
    const float* k  = (const float*)d_in[1];
    const float* v  = (const float*)d_in[2];
    const float* pe = (const float*)d_in[3];
    const float* Wq = (const float*)d_in[4];
    const float* bq = (const float*)d_in[5];
    const float* Wk = (const float*)d_in[6];
    const float* bk = (const float*)d_in[7];
    const float* Wv = (const float*)d_in[8];
    const float* bv = (const float*)d_in[9];
    const float* Wp = (const float*)d_in[10];
    const float* ub = (const float*)d_in[11];
    const float* vb = (const float*)d_in[12];
    const float* Wo = (const float*)d_in[13];
    const float* bo = (const float*)d_in[14];

    const int M = 8 * SEQ;                       // 8192 rows
    const size_t SEG = (size_t)M * DMODEL * 2;   // 16 MiB per bf16 intermediate
    char* ws = (char*)d_ws;
    __bf16* qh  = (__bf16*)(ws);
    __bf16* khb = (__bf16*)(ws + SEG);
    __bf16* vhb = (__bf16*)(ws + 2 * SEG);
    __bf16* phb = (__bf16*)(ws + 3 * SEG);
    __bf16* ctx = (__bf16*)(ws + 4 * SEG);

    dim3 gg(DMODEL / 64, M / 128), bb(256);
    hipLaunchKernelGGL((gemm_xwt_kernel<false, true>), gg, bb, 0, stream,
                       (const void*)q, Wq, bq, (void*)qh, M, DMODEL, DMODEL);
    hipLaunchKernelGGL((gemm_xwt_kernel<false, true>), gg, bb, 0, stream,
                       (const void*)k, Wk, bk, (void*)khb, M, DMODEL, DMODEL);
    hipLaunchKernelGGL((gemm_xwt_kernel<false, true>), gg, bb, 0, stream,
                       (const void*)v, Wv, bv, (void*)vhb, M, DMODEL, DMODEL);
    hipLaunchKernelGGL((gemm_xwt_kernel<false, true>), gg, bb, 0, stream,
                       (const void*)pe, Wp, (const float*)nullptr, (void*)phb,
                       M, DMODEL, DMODEL);

    hipError_t err = hipFuncSetAttribute(
        reinterpret_cast<const void*>(&txl_attn_kernel),
        hipFuncAttributeMaxDynamicSharedMemorySize, ATT_SMEM);
    (void)err;
    hipLaunchKernelGGL(txl_attn_kernel, dim3(SEQ / 64, NHEAD, 8), dim3(256),
                       ATT_SMEM, stream, qh, khb, vhb, phb, ub, vb, ctx);

    hipLaunchKernelGGL((gemm_xwt_kernel<true, false>), gg, bb, 0, stream,
                       (const void*)ctx, Wo, bo, d_out, M, DMODEL, DMODEL);
}
